// Model_24240795419357
// MI455X (gfx1250) — matrix operand order repaired and checked by reading, not compiled, not run on hardware
//
#include <hip/hip_runtime.h>
#include <hip/hip_bf16.h>

#define HH 8
#define DD 32
#define HD 256

typedef __attribute__((ext_vector_type(16))) _Float16 v16h;
typedef __attribute__((ext_vector_type(8)))  _Float16 v8h;
typedef __attribute__((ext_vector_type(4)))  _Float16 v4h;
typedef __attribute__((ext_vector_type(8)))  float    v8f;

// ---------------------------------------------------------------------------
// GEMM: C[M,Nn] = A[M,256] * B[256,Nn], fp32 in/out, f16 WMMA compute.
// Block (256 thr / 8 waves) computes a 32x128 macro-tile. K staged through
// LDS in two 128-wide chunks (As row-major k-contig, Bs column-major
// k-contig so fragments are contiguous ds_load_b128 reads). Each wave
// computes two 16x16 tiles sharing its B fragment.
// ---------------------------------------------------------------------------
__global__ __launch_bounds__(256) void gemm_wmma_k256(
    const float* __restrict__ A, const float* __restrict__ B,
    float* __restrict__ C, int M, int Nn) {
  const int K  = 256;
  const int KC = 128;                       // K chunk staged in LDS
  __shared__ _Float16 As[32 * KC];          // [m][k] k-contiguous      (8 KB)
  __shared__ _Float16 Bs[128 * KC];         // [n][k] k-contiguous     (32 KB)

  const int tid  = threadIdx.x;
  const int lane = tid & 31;
  const int wave = tid >> 5;
  const int half = lane >> 4;
  const int l16  = lane & 15;
  const int mbase = blockIdx.x * 32;
  const int nbase = blockIdx.y * 128;

  v8f c0 = {0.f, 0.f, 0.f, 0.f, 0.f, 0.f, 0.f, 0.f};
  v8f c1 = {0.f, 0.f, 0.f, 0.f, 0.f, 0.f, 0.f, 0.f};

  for (int kbase = 0; kbase < K; kbase += KC) {
    // ---- stage A tile: 32 rows x 128 k (f32 -> f16), float4 reads ----
#pragma unroll
    for (int t = 0; t < 4; ++t) {
      int g  = tid + t * 256;               // 1024 groups of 4 k
      int m  = g >> 5;                      // 32 groups per row
      int kq = (g & 31) * 4;
      float4 av = *(const float4*)(A + (size_t)(mbase + m) * K + kbase + kq);
      v4h h4;
      h4[0] = (_Float16)av.x; h4[1] = (_Float16)av.y;
      h4[2] = (_Float16)av.z; h4[3] = (_Float16)av.w;
      *(v4h*)(&As[m * KC + kq]) = h4;
    }
    // ---- stage B tile transposed: 128 k x 128 n -> Bs[n][k] ----
#pragma unroll
    for (int t = 0; t < 16; ++t) {
      int g  = tid + t * 256;               // 4096 groups of 4 n
      int k  = g >> 5;                      // 32 groups per k-row
      int n4 = (g & 31) * 4;
      float4 bv = *(const float4*)(B + (size_t)(kbase + k) * Nn + nbase + n4);
      Bs[(n4 + 0) * KC + k] = (_Float16)bv.x;
      Bs[(n4 + 1) * KC + k] = (_Float16)bv.y;
      Bs[(n4 + 2) * KC + k] = (_Float16)bv.z;
      Bs[(n4 + 3) * KC + k] = (_Float16)bv.w;
    }
    __syncthreads();

    // ---- 4 WMMA k-steps over this chunk ----
    const int nloc = wave * 16 + l16;       // this lane's column (B/C layout)
#pragma unroll
    for (int kk = 0; kk < KC; kk += 32) {
      // A fragment lane layout: a[0..15] = K kk+16*half+{0..15}
      const _Float16* ap0 = &As[l16 * KC + kk + 16 * half];
      const _Float16* ap1 = &As[(16 + l16) * KC + kk + 16 * half];
      v8h lo0 = *(const v8h*)(ap0);
      v8h hi0 = *(const v8h*)(ap0 + 8);
      v8h lo1 = *(const v8h*)(ap1);
      v8h hi1 = *(const v8h*)(ap1 + 8);
      v16h a0 = __builtin_shufflevector(lo0, hi0, 0,1,2,3,4,5,6,7,8,9,10,11,12,13,14,15);
      v16h a1 = __builtin_shufflevector(lo1, hi1, 0,1,2,3,4,5,6,7,8,9,10,11,12,13,14,15);
      // B fragment: b[0..15] = K kk+16*half+{0..15}, contiguous in Bs[n][k]
      v16h b = *(const v16h*)(&Bs[nloc * KC + kk + 16 * half]);
      c0 = __builtin_amdgcn_wmma_f32_16x16x32_f16(false, a0, false, b,
                                                  (short)0, c0, false, false);
      c1 = __builtin_amdgcn_wmma_f32_16x16x32_f16(false, a1, false, b,
                                                  (short)0, c1, false, false);
    }
    __syncthreads();
  }

  const int n = nbase + wave * 16 + l16;
#pragma unroll
  for (int r = 0; r < 8; ++r) {
    C[(size_t)(mbase + r + 8 * half) * Nn + n]      = c0[r];
    C[(size_t)(mbase + 16 + r + 8 * half) * Nn + n] = c1[r];
  }
}

// ---------------------------------------------------------------------------
// Column mean/var (biased, training-mode BN) over M rows of Y[M,Nn].
// ---------------------------------------------------------------------------
__global__ __launch_bounds__(256) void col_stats(
    const float* __restrict__ Y, float* __restrict__ mean,
    float* __restrict__ var, int M, int Nn) {
  const int j = blockIdx.x;
  float s = 0.f, s2 = 0.f;
  for (int i = threadIdx.x; i < M; i += blockDim.x) {
    float v = Y[(size_t)i * Nn + j];
    s += v; s2 += v * v;
  }
  __shared__ float sh[256], sh2[256];
  sh[threadIdx.x] = s; sh2[threadIdx.x] = s2;
  __syncthreads();
  for (int st = 128; st > 0; st >>= 1) {
    if (threadIdx.x < st) {
      sh[threadIdx.x]  += sh[threadIdx.x + st];
      sh2[threadIdx.x] += sh2[threadIdx.x + st];
    }
    __syncthreads();
  }
  if (threadIdx.x == 0) {
    float mm = sh[0] / (float)M;
    mean[j] = mm;
    var[j]  = sh2[0] / (float)M - mm * mm;
  }
}

// x = leaky(bn(Y), 0.01) + S + short_b
__global__ void proj_combine(const float* __restrict__ Y,
                             const float* __restrict__ S,
                             const float* __restrict__ g,
                             const float* __restrict__ b,
                             const float* __restrict__ shortb,
                             const float* __restrict__ mean,
                             const float* __restrict__ var,
                             float* __restrict__ X, int total) {
  int idx = blockIdx.x * blockDim.x + threadIdx.x;
  if (idx >= total) return;
  int j = idx & (HD - 1);
  float y = Y[idx];
  float bn = g[j] * (y - mean[j]) * rsqrtf(var[j] + 1e-5f) + b[j];
  float act = (bn >= 0.f) ? bn : 0.01f * bn;
  X[idx] = act + S[idx] + shortb[j];
}

// el/er: out[i,h] = sum_d F[i,h,d] * av[h,d]
__global__ void attn_proj(const float* __restrict__ F,
                          const float* __restrict__ av,
                          float* __restrict__ out, int NHtot) {
  int idx = blockIdx.x * blockDim.x + threadIdx.x;
  if (idx >= NHtot) return;
  int i = idx >> 3, h = idx & 7;
  const float* fp = F + (size_t)i * HD + h * DD;
  const float* ap = av + h * DD;
  float s = 0.f;
#pragma unroll
  for (int d = 0; d < DD; ++d) s += fp[d] * ap[d];
  out[idx] = s;
}

// monotonic order-preserving float<->uint mapping for atomicMax
__device__ __forceinline__ unsigned fmap(float f) {
  unsigned u = __float_as_uint(f);
  return (u & 0x80000000u) ? ~u : (u | 0x80000000u);
}
__device__ __forceinline__ float funmap(unsigned u) {
  u = (u & 0x80000000u) ? (u & 0x7FFFFFFFu) : ~u;
  return __uint_as_float(u);
}

__global__ void init_emax(unsigned* __restrict__ emax_u, int n) {
  int idx = blockIdx.x * blockDim.x + threadIdx.x;
  if (idx < n) emax_u[idx] = fmap(-__builtin_inff());
}

// e = leaky(el[src]+er[dst], 0.2); segment max by dst
__global__ void edge_pass1(const int* __restrict__ src, const int* __restrict__ dst,
                           const float* __restrict__ el, const float* __restrict__ er,
                           float* __restrict__ ee, unsigned* __restrict__ emax_u,
                           int EHtot) {
  int idx = blockIdx.x * blockDim.x + threadIdx.x;
  if (idx >= EHtot) return;
  int e = idx >> 3, h = idx & 7;
  float v = el[src[e] * HH + h] + er[dst[e] * HH + h];
  v = (v >= 0.f) ? v : 0.2f * v;
  ee[idx] = v;
  atomicMax(&emax_u[dst[e] * HH + h], fmap(v));
}

__global__ void emax_fin(const unsigned* __restrict__ emax_u,
                         float* __restrict__ emax_f, int n) {
  int idx = blockIdx.x * blockDim.x + threadIdx.x;
  if (idx >= n) return;
  float f = funmap(emax_u[idx]);
  emax_f[idx] = (f >= -3.0e38f && f <= 3.0e38f) ? f : 0.0f;
}

// ee = exp(e - emax[dst]); segment sum
__global__ void edge_pass2(const int* __restrict__ dst, float* __restrict__ ee,
                           const float* __restrict__ emax_f, float* __restrict__ den,
                           int EHtot) {
  int idx = blockIdx.x * blockDim.x + threadIdx.x;
  if (idx >= EHtot) return;
  int e = idx >> 3, h = idx & 7;
  float v = __expf(ee[idx] - emax_f[dst[e] * HH + h]);
  ee[idx] = v;
  atomicAdd(&den[dst[e] * HH + h], v);
}

// out[dst,h,:] += (ee/den) * fs[src,h,:]
__global__ void edge_pass3(const int* __restrict__ src, const int* __restrict__ dst,
                           const float* __restrict__ ee, const float* __restrict__ den,
                           const float* __restrict__ fs, float* __restrict__ out,
                           int EHtot) {
  int idx = blockIdx.x * blockDim.x + threadIdx.x;
  if (idx >= EHtot) return;
  int e = idx >> 3, h = idx & 7;
  int s = src[e], d = dst[e];
  float a = ee[idx] / fmaxf(den[d * HH + h], 1e-9f);
  const float* fp = fs + (size_t)s * HD + h * DD;
  float* op = out + (size_t)d * HD + h * DD;
#pragma unroll
  for (int k = 0; k < DD; ++k) atomicAdd(&op[k], a * fp[k]);
}

__global__ void leaky_ker(float* __restrict__ p, int total, float slope) {
  int idx = blockIdx.x * blockDim.x + threadIdx.x;
  if (idx >= total) return;
  float v = p[idx];
  p[idx] = (v >= 0.f) ? v : slope * v;
}

// acc += sum_{i,j} tanh(T[i,j] + b1[j]) * w2[j]
__global__ __launch_bounds__(256) void sem_reduce(
    const float* __restrict__ T, const float* __restrict__ b1,
    const float* __restrict__ w2, float* __restrict__ acc, int total) {
  __shared__ float sh[256];
  float s = 0.f;
  for (int idx = blockIdx.x * blockDim.x + threadIdx.x; idx < total;
       idx += gridDim.x * blockDim.x) {
    int j = idx & 127;
    s += tanhf(T[idx] + b1[j]) * w2[j];
  }
  sh[threadIdx.x] = s;
  __syncthreads();
  for (int st = 128; st > 0; st >>= 1) {
    if (threadIdx.x < st) sh[threadIdx.x] += sh[threadIdx.x + st];
    __syncthreads();
  }
  if (threadIdx.x == 0) atomicAdd(acc, sh[0]);
}

// wacc[0..3] = raw sums for channels {e1,e2,e0,e3}; write betas to wacc[4..7]
__global__ void beta_ker(float* __restrict__ wacc, int Nd) {
  float w0 = wacc[0] / (float)Nd, w1 = wacc[1] / (float)Nd;  // l: e1,e2
  float m = fmaxf(w0, w1);
  float a0 = __expf(w0 - m), a1 = __expf(w1 - m);
  wacc[4] = a0 / (a0 + a1);
  wacc[5] = a1 / (a0 + a1);
  float v0 = wacc[2] / (float)Nd, v1 = wacc[3] / (float)Nd;  // d: e0,e3
  float md = fmaxf(v0, v1);
  float b0 = __expf(v0 - md), b1 = __expf(v1 - md);
  wacc[6] = b0 / (b0 + b1);
  wacc[7] = b1 / (b0 + b1);
}

// l_embs = beta_l0*e1 + beta_l1*e2 ; d_embs = beta_d0*e0 + beta_d1*e3
__global__ void combine_embs(const float* __restrict__ eacc,
                             const float* __restrict__ wacc,
                             float* __restrict__ lout, float* __restrict__ dout,
                             int nm) {
  int idx = blockIdx.x * blockDim.x + threadIdx.x;
  if (idx >= nm) return;
  size_t NM = (size_t)nm;
  lout[idx] = wacc[4] * eacc[NM + idx]     + wacc[5] * eacc[2 * NM + idx];
  dout[idx] = wacc[6] * eacc[idx]          + wacc[7] * eacc[3 * NM + idx];
}

__global__ void edge_mul(const int* __restrict__ src, const int* __restrict__ dst,
                         const float* __restrict__ l, const float* __restrict__ d,
                         float* __restrict__ out, int total) {
  int idx = blockIdx.x * blockDim.x + threadIdx.x;
  if (idx >= total) return;
  int e = idx >> 8, j = idx & 255;
  out[idx] = l[(size_t)src[e] * HD + j] * d[(size_t)dst[e] * HD + j];
}

// ---------------------------------------------------------------------------
extern "C" void kernel_launch(void* const* d_in, const int* in_sizes, int n_in,
                              void* d_out, int out_size, void* d_ws, size_t ws_size,
                              hipStream_t stream) {
  (void)n_in; (void)out_size; (void)ws_size;
  const float* x_lnc   = (const float*)d_in[0];
  const float* x_dis   = (const float*)d_in[1];
  const float* proj_W  = (const float*)d_in[2];
  const float* bn_g    = (const float*)d_in[3];
  const float* bn_b    = (const float*)d_in[4];
  const float* short_W = (const float*)d_in[5];
  const float* short_b = (const float*)d_in[6];
  const float* gat_W   = (const float*)d_in[7];
  const float* gat_al  = (const float*)d_in[8];
  const float* gat_ar  = (const float*)d_in[9];
  const float* sem_W1  = (const float*)d_in[10];
  const float* sem_b1  = (const float*)d_in[11];
  const float* sem_w2  = (const float*)d_in[12];
  const int* gsrc[4] = {(const int*)d_in[13], (const int*)d_in[15],
                        (const int*)d_in[17], (const int*)d_in[19]};
  const int* gdst[4] = {(const int*)d_in[14], (const int*)d_in[16],
                        (const int*)d_in[18], (const int*)d_in[20]};
  const int* pos_src = (const int*)d_in[21];
  const int* pos_dst = (const int*)d_in[22];
  const int* neg_src = (const int*)d_in[23];
  const int* neg_dst = (const int*)d_in[24];

  const int Nd = in_sizes[0] / HD;   // 20000
  const int E  = in_sizes[13];       // 160000
  const size_t NM = (size_t)Nd * HD;
  const size_t EH = (size_t)E * HH;
  const size_t NH = (size_t)Nd * HH;

  float* ws    = (float*)d_ws;
  float* l_x   = ws;
  float* d_x   = ws + NM;
  float* eacc  = ws + 2 * NM;           // 4 * NM (GAT outputs e0..e3)
  float* fb0   = ws + 6 * NM;           // fs / y / T
  float* fb1   = ws + 7 * NM;           // fd / shortcut
  float* eeb   = ws + 8 * NM;           // E*H edge scratch
  unsigned* emax_u = (unsigned*)(eeb + EH);
  float* emax_f = (float*)emax_u + NH;
  float* den    = emax_f + NH;
  float* elb    = den + NH;
  float* erb    = elb + NH;
  float* meanb  = erb + NH;
  float* varb   = meanb + HD;
  float* wacc   = varb + HD;            // [0..3] sums, [4..7] betas

  const dim3 blk(256);
  const int mt = Nd / 32;               // 625 macro-tiles of 32 rows
  const int nb_nm = (int)((NM + 255) / 256);
  const int nb_nh = (int)((NH + 255) / 256);
  const int nb_eh = (int)((EH + 255) / 256);

  hipMemsetAsync(eacc, 0, 4 * NM * sizeof(float), stream);
  hipMemsetAsync(wacc, 0, 8 * sizeof(float), stream);

  // ---- ResNet projection, branch lnc ----
  gemm_wmma_k256<<<dim3(mt, 2), blk, 0, stream>>>(x_lnc, proj_W, fb0, Nd, HD);
  gemm_wmma_k256<<<dim3(mt, 2), blk, 0, stream>>>(x_lnc, short_W, fb1, Nd, HD);
  col_stats<<<HD, blk, 0, stream>>>(fb0, meanb, varb, Nd, HD);
  proj_combine<<<nb_nm, blk, 0, stream>>>(fb0, fb1, bn_g, bn_b, short_b,
                                          meanb, varb, l_x, (int)NM);
  // ---- branch dis ----
  gemm_wmma_k256<<<dim3(mt, 2), blk, 0, stream>>>(x_dis, proj_W + 65536, fb0, Nd, HD);
  gemm_wmma_k256<<<dim3(mt, 2), blk, 0, stream>>>(x_dis, short_W + 65536, fb1, Nd, HD);
  col_stats<<<HD, blk, 0, stream>>>(fb0, meanb, varb, Nd, HD);
  proj_combine<<<nb_nm, blk, 0, stream>>>(fb0, fb1, bn_g, bn_b, short_b + HD,
                                          meanb, varb, d_x, (int)NM);

  // ---- 4 meta-path GATs: (l,d) (l,l) (d,l) (d,d) ----
  for (int g = 0; g < 4; ++g) {
    const float* hs = (g == 0 || g == 1) ? l_x : d_x;
    const float* hd = (g == 1 || g == 2) ? l_x : d_x;
    const float* Wg = gat_W + (size_t)g * 65536;
    gemm_wmma_k256<<<dim3(mt, 2), blk, 0, stream>>>(hs, Wg, fb0, Nd, HD);
    const float* fdp = fb0;
    if (hs != hd) {
      gemm_wmma_k256<<<dim3(mt, 2), blk, 0, stream>>>(hd, Wg, fb1, Nd, HD);
      fdp = fb1;
    }
    attn_proj<<<nb_nh, blk, 0, stream>>>(fb0, gat_al + g * HD, elb, (int)NH);
    attn_proj<<<nb_nh, blk, 0, stream>>>(fdp, gat_ar + g * HD, erb, (int)NH);
    init_emax<<<nb_nh, blk, 0, stream>>>(emax_u, (int)NH);
    hipMemsetAsync(den, 0, NH * sizeof(float), stream);
    edge_pass1<<<nb_eh, blk, 0, stream>>>(gsrc[g], gdst[g], elb, erb, eeb,
                                          emax_u, (int)EH);
    emax_fin<<<nb_nh, blk, 0, stream>>>(emax_u, emax_f, (int)NH);
    edge_pass2<<<nb_eh, blk, 0, stream>>>(gdst[g], eeb, emax_f, den, (int)EH);
    edge_pass3<<<nb_eh, blk, 0, stream>>>(gsrc[g], gdst[g], eeb, den, fb0,
                                          eacc + (size_t)g * NM, (int)EH);
  }
  leaky_ker<<<(int)((4 * NM + 255) / 256), blk, 0, stream>>>(eacc, (int)(4 * NM), 0.01f);

  // ---- semantic attention: channels e1,e2 (lnc) then e0,e3 (dis) ----
  const float* chans[4] = {eacc + NM, eacc + 2 * NM, eacc, eacc + 3 * NM};
  for (int c = 0; c < 4; ++c) {
    gemm_wmma_k256<<<dim3(mt, 1), blk, 0, stream>>>(chans[c], sem_W1, fb0, Nd, 128);
    sem_reduce<<<512, blk, 0, stream>>>(fb0, sem_b1, sem_w2, wacc + c,
                                        (int)((size_t)Nd * 128));
  }
  beta_ker<<<1, 1, 0, stream>>>(wacc, Nd);

  float* out = (float*)d_out;
  float* lout = out + 2 * (size_t)E * HD;   // l_embs
  float* dout = lout + NM;                  // d_embs
  combine_embs<<<nb_nm, blk, 0, stream>>>(eacc, wacc, lout, dout, (int)NM);

  const int etot = E * HD;
  const int nb_e = (etot + 255) / 256;
  edge_mul<<<nb_e, blk, 0, stream>>>(pos_src, pos_dst, lout, dout, out, etot);
  edge_mul<<<nb_e, blk, 0, stream>>>(neg_src, neg_dst, lout, dout,
                                     out + (size_t)E * HD, etot);
}
